// EruSelfAttentionModel_13322988552231
// MI455X (gfx1250) — matrix operand order repaired and checked by reading, not compiled, not run on hardware
//
#include <hip/hip_runtime.h>
#include <hip/hip_bf16.h>
#include <cstdint>
#include <math.h>

// ---------------------------------------------------------------------------
// EruSelfAttention for MI455X (gfx1250, wave32, WMMA + TDM).
//
// out = blend_weights(QK^T/sqrt(E)) @ (h @ Wv^T)
//     = (blend_weights(...) @ h) @ Wv^T        <- V factored out (saves 64MB)
//
// Matmuls: v_wmma_f32_16x16x32_f16 (f16 A/B, fp32 accumulate).
// LDS staging in the attention kernel: Tensor Data Mover (tensor_load_to_lds),
// one descriptor per tile, waited with s_wait_tensorcnt.
// ---------------------------------------------------------------------------

typedef __attribute__((ext_vector_type(16))) _Float16 v16h;
typedef __attribute__((ext_vector_type(8)))  float    v8f;
typedef __attribute__((ext_vector_type(4)))  uint32_t u32x4;
typedef __attribute__((ext_vector_type(8)))  int32_t  i32x8;
typedef __attribute__((ext_vector_type(4)))  int32_t  i32x4;
typedef _Float16 h16;

#define EMB   512
#define ADIM  64
#define HEADS 8
#define BB    8
#define SS    1024

// A-fragment (16-bit 16x32): lane m = lane&15; half j -> K
__device__ __forceinline__ int aK(int lane, int j) {
  return ((lane >> 4) << 3) + j + ((j >> 3) << 3);
}
// B-fragment (16-bit 32x16): lane n = lane&15; half j -> K
__device__ __forceinline__ int bK(int lane, int j) {
  return ((lane >> 4) << 3) + j + ((j >> 3) << 3);
}

// ---------------------------------------------------------------------------
// Tensor Data Mover: load a 2-D f16 tile (rows x row_elems, global row stride
// row_stride elems) into contiguous LDS at lds_dst.  D# layout per CDNA5 ISA
// ch.8 (group0: count/lds/global/type; group1: data_size, dims, tile, stride).
// The low 32 bits of a generic LDS pointer are the LDS byte offset.
// ---------------------------------------------------------------------------
__device__ __forceinline__ void tdm_load_2d_f16(
    void* lds_dst, const h16* gsrc,
    uint32_t row_elems, uint32_t rows, uint32_t row_stride)
{
  const uint64_t ga = (uint64_t)(uintptr_t)gsrc;
  u32x4 g0;
  g0[0] = 1u;                                   // count=1, user descriptor
  g0[1] = (uint32_t)(uintptr_t)lds_dst;         // lds_addr (bytes)
  g0[2] = (uint32_t)ga;                         // global_addr[31:0]
  g0[3] = (uint32_t)(ga >> 32) | (2u << 30);    // global_addr[56:32] | type=2
  i32x8 g1;
  g1[0] = (int)(1u << 16);                      // data_size=1 -> 2 bytes
  g1[1] = (int)((row_elems & 0xFFFFu) << 16);   // tensor_dim0[15:0]
  g1[2] = (int)((row_elems >> 16) | ((rows & 0xFFFFu) << 16)); // dim0 hi|dim1 lo
  g1[3] = (int)((rows >> 16) | ((row_elems & 0xFFFFu) << 16)); // dim1 hi|tile_dim0
  g1[4] = (int)rows;                            // tile_dim1 (tile_dim2=0)
  g1[5] = (int)row_stride;                      // tensor_dim0_stride[31:0]
  g1[6] = 0;                                    // stride hi | dim1_stride lo
  g1[7] = 0;
  i32x4 z4 = {0, 0, 0, 0};                      // groups 2/3 unused (2-D tile)
#if defined(__clang_major__) && __clang_major__ >= 23
  i32x8 z8 = {0, 0, 0, 0, 0, 0, 0, 0};
  __builtin_amdgcn_tensor_load_to_lds(g0, g1, z4, z4, z8, 0);
#else
  __builtin_amdgcn_tensor_load_to_lds(g0, g1, z4, z4, 0);
#endif
}

// ---------------------------------------------------------------------------
// Kernel 1: embedding gather + LayerNorm, emit h as f16.  One wave32 per row.
// ---------------------------------------------------------------------------
__global__ __launch_bounds__(256) void embed_ln_kernel(
    const int* __restrict__ x, const float* __restrict__ emb,
    const float* __restrict__ gamma, const float* __restrict__ beta,
    h16* __restrict__ hOut)
{
  const int lane = threadIdx.x & 31;
  const int wave = threadIdx.x >> 5;
  const int row  = blockIdx.x * 8 + wave;          // B*S = 8192 rows
  const int tok  = x[row];
  const float* e = emb + (long)tok * EMB;

  float v[16], s = 0.f, s2 = 0.f;
#pragma unroll
  for (int i = 0; i < 16; ++i) {
    float t = e[lane + i * 32];                    // coalesced
    v[i] = t; s += t; s2 += t * t;
  }
#pragma unroll
  for (int m = 16; m >= 1; m >>= 1) {              // wave32 butterfly reduce
    s  += __shfl_xor(s,  m, 32);
    s2 += __shfl_xor(s2, m, 32);
  }
  const float mu   = s * (1.f / EMB);
  const float var  = s2 * (1.f / EMB) - mu * mu;
  const float rstd = rsqrtf(var + 1e-5f);

  h16* dst = hOut + (long)row * EMB;
#pragma unroll
  for (int i = 0; i < 16; ++i) {
    int c = lane + i * 32;
    dst[c] = (h16)((v[i] - mu) * rstd * gamma[c] + beta[c]);
  }
}

// ---------------------------------------------------------------------------
// Kernel 2: f32 -> f16 weight conversion (grid-stride).
// ---------------------------------------------------------------------------
__global__ __launch_bounds__(256) void f32_to_f16_kernel(
    const float* __restrict__ src, h16* __restrict__ dst, int n)
{
  int i = blockIdx.x * blockDim.x + threadIdx.x;
  int stride = gridDim.x * blockDim.x;
  for (; i < n; i += stride) dst[i] = (h16)src[i];
}

// ---------------------------------------------------------------------------
// Kernel 3: Q/K projection.  Per batch b: Out[b] = h[b](1024x512) x W^T,
// where W is Wq or Wk viewed as [H*A=512, E=512].  Block = 4 waves; each
// block does a 16x64 output tile; each wave one 16x16 tile.
// grid = (B*S/16, (H*A)/64, 2 {q,k})
// ---------------------------------------------------------------------------
__global__ __launch_bounds__(128) void qk_proj_kernel(
    const h16* __restrict__ hF16, const h16* __restrict__ WqF16,
    const h16* __restrict__ WkF16, h16* __restrict__ Qf16,
    h16* __restrict__ Kf16)
{
  __shared__ __attribute__((aligned(16))) h16 Atile[16 * 32];
  __shared__ __attribute__((aligned(16))) h16 Btile[64 * 32];

  const int tid  = threadIdx.x;
  const int lane = tid & 31;
  const int wave = tid >> 5;                 // 0..3
  const int b    = blockIdx.x >> 6;          // 64 row-tiles per batch
  const int m0   = (blockIdx.x & 63) * 16;
  const int c0   = blockIdx.y * 64;          // output-column base
  const int laneM = lane & 15;

  const h16* W   = (blockIdx.z == 0) ? WqF16 : WkF16;
  h16*       Out = (blockIdx.z == 0) ? Qf16  : Kf16;
  const h16* hB  = hF16 + (long)b * SS * EMB;

  v8f acc = {};
  for (int ks = 0; ks < 16; ++ks) {          // K = 512 in steps of 32
    const int k0 = ks * 32;
    __syncthreads();
    { // stage A: 16 rows x 32 halves; each thread 4 halves (8B)
      int r  = tid >> 3;
      int ch = tid & 7;
      *(uint64_t*)(Atile + r * 32 + ch * 4) =
          *(const uint64_t*)(hB + (long)(m0 + r) * EMB + k0 + ch * 4);
    }
    { // stage B: 64 weight rows x 32 halves; each thread 8 halves (16B)
      int r  = tid >> 1;
      int ch = tid & 1;
      *(uint4*)(Btile + r * 32 + ch * 16) =
          *(const uint4*)(W + (long)(c0 + r) * EMB + k0 + ch * 16);
    }
    __syncthreads();

    v16h a, bf;
#pragma unroll
    for (int j = 0; j < 16; ++j) a[j]  = Atile[laneM * 32 + aK(lane, j)];
#pragma unroll
    for (int j = 0; j < 16; ++j) bf[j] = Btile[(wave * 16 + laneM) * 32 + bK(lane, j)];
    acc = __builtin_amdgcn_wmma_f32_16x16x32_f16(false, a, false, bf,
                                                 (short)0, acc, false, false);
  }

  // scatter C tile into [B,H,S,A] f16
#pragma unroll
  for (int r = 0; r < 8; ++r) {
    int m    = ((lane >> 4) << 3) + r;
    int c    = c0 + wave * 16 + laneM;
    int head = c >> 6, ad = c & 63;
    Out[((long)(b * HEADS + head) * SS + (m0 + m)) * ADIM + ad] = (h16)acc[r];
  }
}

// ---------------------------------------------------------------------------
// Kernel 4: attention.  One workgroup (8 waves) per (b, head, 16-row stripe).
// All bulk LDS staging via the Tensor Data Mover (wave 0 issues, waits
// TENSORcnt, block barrier publishes).
// LDS (dynamic, ~196KB of the 320KB WGP pool):
//   [0      .. 131072)  K tile 1024x64 f16 (phase1) / h & Wv staging + wh
//   [131072 .. 196608)  scores/weights 16x1024 f32
//   [196608 .. 198656)  Q tile 16x64 f16
//   [198656 .. 200832)  reduction scratch
// ---------------------------------------------------------------------------
#define ATTN_SMEM_BYTES 200960

__global__ __launch_bounds__(256) void attn_kernel(
    const h16* __restrict__ hF16, const h16* __restrict__ Qf16,
    const h16* __restrict__ Kf16, const h16* __restrict__ WvF16,
    float* __restrict__ out)
{
  extern __shared__ char smem[];
  h16*   Kls    = (h16*)smem;                         // phase 1
  h16*   stage  = (h16*)smem;                         // phase 3/4 (32 KB)
  h16*   whLDS  = (h16*)(smem + 32768);               // 16x512 f16
  float* scores = (float*)(smem + 131072);            // 16x1024 f32
  h16*   Qls    = (h16*)(smem + 131072 + 65536);
  float* redA   = (float*)(smem + 198656);            // 256 floats
  float* redB   = (float*)(smem + 198656 + 1024);     // 256 floats
  float* rowA   = (float*)(smem + 198656 + 2048);     // 16 floats
  float* rowB   = (float*)(smem + 198656 + 2048 + 64);// 16 floats

  const int tid   = threadIdx.x;
  const int lane  = tid & 31;
  const int wave  = tid >> 5;                  // 0..7
  const int laneM = lane & 15;
  const int m0    = blockIdx.x * 16;
  const int head  = blockIdx.y;
  const int b     = blockIdx.z;
  const int bh    = b * HEADS + head;

  // ---- stage Q (16x64, vmem) and all of K (1024x64, TDM) into LDS ----
  if (wave == 0)   // one DMA descriptor moves the whole 128KB K tile
    tdm_load_2d_f16(Kls, Kf16 + (long)bh * SS * ADIM, ADIM, SS, ADIM);
  {
    const uint4* qs = (const uint4*)(Qf16 + ((long)bh * SS + m0) * ADIM);
    if (tid < 128) ((uint4*)Qls)[tid] = qs[tid];
  }
  if (wave == 0) __builtin_amdgcn_s_wait_tensorcnt(0);
  __syncthreads();

  // ---- phase 1: scores = Q K^T / sqrt(EMB) ----------------------------------
  const float scale = 0.044194173824159216f;   // 1/sqrt(512) (ref uses sqrt(EMB))
  for (int i = 0; i < 8; ++i) {                // 64 col-tiles / 8 waves
    const int n0 = (wave + i * 8) * 16;
    v8f c = {};
#pragma unroll
    for (int kk = 0; kk < 2; ++kk) {           // A-dim 64 = 2 k-steps
      const int k0 = kk * 32;
      v16h a, bf;
#pragma unroll
      for (int j = 0; j < 16; ++j) a[j]  = Qls[laneM * ADIM + k0 + aK(lane, j)];
#pragma unroll
      for (int j = 0; j < 16; ++j) bf[j] = Kls[(n0 + laneM) * ADIM + k0 + bK(lane, j)];
      c = __builtin_amdgcn_wmma_f32_16x16x32_f16(false, a, false, bf,
                                                 (short)0, c, false, false);
    }
#pragma unroll
    for (int r = 0; r < 8; ++r) {
      int m = ((lane >> 4) << 3) + r;
      scores[m * SS + n0 + laneM] = c[r] * scale;
    }
  }
  __syncthreads();

  // ---- phase 2: min-max normalize + softmax/sigmoid blend -------------------
  const int row = tid >> 4;                    // 16 threads per row
  const int sub = tid & 15;
  float mx = -3.0e38f, mn = 3.0e38f;
  for (int c = sub; c < SS; c += 16) {
    float v = scores[row * SS + c];
    mx = fmaxf(mx, v); mn = fminf(mn, v);
  }
  redA[tid] = mx; redB[tid] = mn;
  __syncthreads();
  if (sub == 0) {
    float m1 = -3.0e38f, m2 = 3.0e38f;
    for (int t = 0; t < 16; ++t) {
      m1 = fmaxf(m1, redA[row * 16 + t]);
      m2 = fminf(m2, redB[row * 16 + t]);
    }
    rowA[row] = m1; rowB[row] = m2;
  }
  __syncthreads();
  const float rmin = rowB[row];
  const float inv  = 1.0f / (rowA[row] - rmin);

  float se = 0.f, sm = -3.0e38f;               // sumexp(sn), max(sn)
  for (int c = sub; c < SS; c += 16) {
    float sn = (scores[row * SS + c] - rmin) * inv;
    se += __expf(sn); sm = fmaxf(sm, sn);
  }
  redA[tid] = sm; redB[tid] = se;
  __syncthreads();
  if (sub == 0) {
    float m1 = -3.0e38f, s1 = 0.f;
    for (int t = 0; t < 16; ++t) {
      m1 = fmaxf(m1, redA[row * 16 + t]);
      s1 += redB[row * 16 + t];
    }
    rowA[row] = m1; rowB[row] = s1;
  }
  __syncthreads();
  const float snmax   = rowA[row];
  const float softsel = 1.0f - snmax;          // == 0 mathematically
  const float sigsel  = snmax;
  const float invSum  = 1.0f / rowB[row];
  for (int c = sub; c < SS; c += 16) {
    float sn = (scores[row * SS + c] - rmin) * inv;
    float w  = softsel * __expf(sn) * invSum
             + sigsel  / (1.0f + __expf(5.0f - 10.0f * sn));
    scores[row * SS + c] = w;                  // weights, in place (f32)
  }
  __syncthreads();

  // ---- phase 3: wh = weights(16x1024) @ h(1024x512) -------------------------
  v8f zacc = {};
  v8f wh[4] = {zacc, zacc, zacc, zacc};
  const h16* hB = hF16 + (long)b * SS * EMB;
  for (int ks = 0; ks < 32; ++ks) {            // t-dim 1024 in steps of 32
    __syncthreads();                           // previous chunk fully consumed
    if (wave == 0) {                           // TDM: 32 rows x 512 f16 (32KB)
      tdm_load_2d_f16(stage, hB + (long)ks * 32 * EMB, EMB, 32, EMB);
      __builtin_amdgcn_s_wait_tensorcnt(0);
    }
    if (ks + 1 < 32)                           // warm L2 for the next chunk
      __builtin_prefetch(hB + (long)(ks + 1) * 32 * EMB + tid * 64, 0, 1);
    __syncthreads();

    v16h a, bf;
#pragma unroll
    for (int j = 0; j < 16; ++j)
      a[j] = (h16)scores[laneM * SS + ks * 32 + aK(lane, j)];
#pragma unroll
    for (int i = 0; i < 4; ++i) {              // 32 col-tiles / 8 waves
      const int n0 = (wave * 4 + i) * 16;
#pragma unroll
      for (int j = 0; j < 16; ++j) bf[j] = stage[bK(lane, j) * EMB + n0 + laneM];
      wh[i] = __builtin_amdgcn_wmma_f32_16x16x32_f16(false, a, false, bf,
                                                     (short)0, wh[i], false, false);
    }
  }
  __syncthreads();
#pragma unroll
  for (int i = 0; i < 4; ++i) {                // spill wh to LDS as f16
    const int n0 = (wave * 4 + i) * 16;
#pragma unroll
    for (int r = 0; r < 8; ++r) {
      int m = ((lane >> 4) << 3) + r;
      whLDS[m * EMB + n0 + laneM] = (h16)wh[i][r];
    }
  }
  __syncthreads();

  // ---- phase 4: out = wh(16x512) @ Wv[head]^T(512x512) ----------------------
  v8f oc[4] = {zacc, zacc, zacc, zacc};
  const h16* Wv = WvF16 + (long)head * EMB * EMB;
  for (int ks = 0; ks < 16; ++ks) {            // e-dim 512 in steps of 32
    __syncthreads();
    if (wave == 0) {                           // TDM: 512 rows x 32 f16, stride 512
      tdm_load_2d_f16(stage, Wv + (long)ks * 32, 32, EMB, EMB);
      __builtin_amdgcn_s_wait_tensorcnt(0);
    }
    __syncthreads();

    v16h a, bf;
#pragma unroll
    for (int j = 0; j < 16; ++j) a[j] = whLDS[laneM * EMB + ks * 32 + aK(lane, j)];
#pragma unroll
    for (int i = 0; i < 4; ++i) {
      const int n0 = (wave * 4 + i) * 16;
#pragma unroll
      for (int j = 0; j < 16; ++j) bf[j] = stage[(n0 + laneM) * 32 + bK(lane, j)];
      oc[i] = __builtin_amdgcn_wmma_f32_16x16x32_f16(false, a, false, bf,
                                                     (short)0, oc[i], false, false);
    }
  }

  // ---- store out fp32 [B,H,S,EMB] ----
  float* ob = out + ((long)bh * SS + m0) * EMB;
#pragma unroll
  for (int i = 0; i < 4; ++i) {
    const int n0 = (wave * 4 + i) * 16;
#pragma unroll
    for (int r = 0; r < 8; ++r) {
      int m = ((lane >> 4) << 3) + r;
      ob[(long)m * EMB + n0 + laneM] = oc[i][r];
    }
  }
}

// ---------------------------------------------------------------------------
// Host launcher.  Workspace layout (bytes), ~29 MB total:
//   hF16  @ 0         8 MB   WqF16 @ 8388608   .5 MB   WkF16 @ 8912896  .5 MB
//   WvF16 @ 9437184   4 MB   Qf16  @ 13631488  8 MB    Kf16  @ 22020096 8 MB
// ---------------------------------------------------------------------------
extern "C" void kernel_launch(void* const* d_in, const int* in_sizes, int n_in,
                              void* d_out, int out_size, void* d_ws, size_t ws_size,
                              hipStream_t stream)
{
  (void)in_sizes; (void)n_in; (void)out_size; (void)ws_size;
  const int*   x     = (const int*)d_in[0];
  const float* emb   = (const float*)d_in[1];
  const float* gamma = (const float*)d_in[2];
  const float* beta  = (const float*)d_in[3];
  const float* Wq    = (const float*)d_in[4];
  const float* Wk    = (const float*)d_in[5];
  const float* Wv    = (const float*)d_in[6];
  float* out = (float*)d_out;

  char* ws = (char*)d_ws;
  h16* hF16  = (h16*)(ws);
  h16* WqF16 = (h16*)(ws + 8388608);
  h16* WkF16 = (h16*)(ws + 8912896);
  h16* WvF16 = (h16*)(ws + 9437184);
  h16* Qf16  = (h16*)(ws + 13631488);
  h16* Kf16  = (h16*)(ws + 22020096);

  embed_ln_kernel<<<dim3(1024), dim3(256), 0, stream>>>(x, emb, gamma, beta, hF16);
  f32_to_f16_kernel<<<dim3(256),  dim3(256), 0, stream>>>(Wq, WqF16, HEADS * ADIM * EMB);
  f32_to_f16_kernel<<<dim3(256),  dim3(256), 0, stream>>>(Wk, WkF16, HEADS * ADIM * EMB);
  f32_to_f16_kernel<<<dim3(2048), dim3(256), 0, stream>>>(Wv, WvF16, HEADS * EMB * EMB);

  qk_proj_kernel<<<dim3(512, 8, 2), dim3(128), 0, stream>>>(hF16, WqF16, WkF16, Qf16, Kf16);

  // >64KB dynamic LDS needs the attribute bumped (idempotent, not stream-ordered)
  hipFuncSetAttribute(reinterpret_cast<const void*>(attn_kernel),
                      hipFuncAttributeMaxDynamicSharedMemorySize, ATTN_SMEM_BYTES);
  attn_kernel<<<dim3(SS / 16, HEADS, BB), dim3(256), ATTN_SMEM_BYTES, stream>>>(
      hF16, Qf16, Kf16, WvF16, out);
}
